// Cubify_2259152797816
// MI455X (gfx1250) — compile-verified, matrix operand order checked by reading, not run on hardware
//
#include <hip/hip_runtime.h>

typedef unsigned int u32;
typedef unsigned char u8;
typedef int v8i __attribute__((ext_vector_type(8)));
typedef int v4i __attribute__((ext_vector_type(4)));

#define NB 2
#define GD 64
#define SLAB 4096            // 64*64
#define L 65
#define LL 4225              // 65*65
#define LSIZE 274625         // 65^3
#define WPB 8583             // ceil(LSIZE/32)
#define TW (NB*WPB)          // 17166

// ---- workspace layout (u32 word offsets) ----
#define W_MASK      0                         // 131072 words (u8[524288] view)
#define W_SLABCNT   131072                    // u32[768]  per (b,c,z) quad counts
#define W_SLABBASE  131840                    // u32[769]  exclusive scan (+ total)
#define W_QPERB     132609                    // u32[2]    WMMA per-batch quad totals
#define W_VPERB     132611                    // u32[3]    v/batch0, v/batch1, Nv
#define W_VBITMAP   132624                    // u32[TW]   vertex lattice bitmap
#define W_WORDPREF  (W_VBITMAP + TW)          // u32[TW+1] vertex rank prefix
#define W_EWORDPREF (W_WORDPREF + TW + 1)     // u32[TW+1] edge rank prefix
#define W_EDGEMASK  (W_EWORDPREF + TW + 1)    // u32[NB*LSIZE] 27-bit neighbor masks
#define W_QUADS     (W_EDGEMASK + NB*LSIZE)   // u32[3145728] packed quads
#define ZTOT        (TW + NB*LSIZE + 2)

#if defined(__has_builtin)
#if __has_builtin(__builtin_amdgcn_global_load_async_to_lds_b128)
#define HAVE_ASYNC_LDS 1
#endif
#if __has_builtin(__builtin_amdgcn_s_wait_asynccnt)
#define HAVE_WAIT_ASYNC 1
#endif
#endif
#ifndef HAVE_ASYNC_LDS
#warning "gfx1250 async-to-LDS builtin not found: falling back to VGPR staging"
#endif

typedef __attribute__((address_space(1))) v4i* gv4i_p;
typedef __attribute__((address_space(3))) v4i* lv4i_p;

// corner offsets (dz<<2 | dy<<1 | dx) per direction c, matching _D rows
__constant__ u8 cornerTab[6][4] = {
    {0x0, 0x1, 0x2, 0x3},   // c0: z-0.5 face
    {0x4, 0x5, 0x6, 0x7},   // c1: z+0.5 face
    {0x4, 0x5, 0x0, 0x1},   // c2: y-0.5 face, mixed z order
    {0x2, 0x3, 0x6, 0x7},   // c3: y+0.5 face
    {0x4, 0x0, 0x6, 0x2},   // c4: x-0.5 face
    {0x1, 0x5, 0x3, 0x7},   // c5: x+0.5 face
};

// ---------------------------------------------------------------- zero scratch
__global__ void k_zero(u32* __restrict__ ws) {
  const u32 i = blockIdx.x * 256u + threadIdx.x;
  if (i < TW)                    ws[W_VBITMAP + i] = 0;
  else if (i < TW + NB*LSIZE)    ws[W_EDGEMASK + (i - TW)] = 0;
  else if (i < ZTOT)             ws[W_QPERB + (i - TW - NB*LSIZE)] = 0;
}

// ------------------------------------------------- stencil: masks + slab counts
__global__ void k_stencil(const float* __restrict__ t, u32* __restrict__ ws) {
  __shared__ float sm[3 * SLAB];       // z-1, z, z+1 slabs (48 KB of 320 KB WGP LDS)
  __shared__ u32 cnt[6];
  const int bz = blockIdx.x;
  const int b = bz >> 6, z = bz & 63;
  const int tid = threadIdx.x;
  if (tid < 6) cnt[tid] = 0;

  #pragma unroll
  for (int s = 0; s < 3; ++s) {
    const int zz = z - 1 + s;
    if ((unsigned)zz < 64u) {
      const float* src = t + (size_t)(b * GD + zz) * SLAB;
      #pragma unroll
      for (int c4 = 0; c4 < 4; ++c4) {
        const int idx = c4 * 1024 + tid * 4;
#if defined(HAVE_ASYNC_LDS)
        // CDNA5 async path: HBM -> LDS without VGPR round-trip
        __builtin_amdgcn_global_load_async_to_lds_b128(
            (gv4i_p)(src + idx),
            (lv4i_p)(&sm[s * SLAB + idx]),
            0, 0);
#else
        *(float4*)(&sm[s * SLAB + idx]) = *(const float4*)(src + idx);
#endif
      }
    } else {
      #pragma unroll
      for (int c4 = 0; c4 < 4; ++c4) {
        const int idx = c4 * 1024 + tid * 4;
        *(float4*)(&sm[s * SLAB + idx]) = make_float4(0.f, 0.f, 0.f, 0.f);
      }
    }
  }
#if defined(HAVE_ASYNC_LDS)
#if defined(HAVE_WAIT_ASYNC)
  __builtin_amdgcn_s_wait_asynccnt(0);
#else
  asm volatile("s_wait_asynccnt 0" ::: "memory");
#endif
#endif
  __syncthreads();

  u8* mask = (u8*)ws;
  const size_t mb = (size_t)(b * GD + z) * SLAB;
  u32 lc0 = 0, lc1 = 0, lc2 = 0, lc3 = 0, lc4 = 0, lc5 = 0;
  for (int j = 0; j < 16; ++j) {
    const int k = tid + j * 256;
    const int y = k >> 6, x = k & 63;
    u32 m = 0;
    if (sm[SLAB + k] > 0.5f) {
      const bool zm = sm[k] > 0.5f;
      const bool zp = sm[2 * SLAB + k] > 0.5f;
      const bool yp = (y < 63) && (sm[SLAB + k + 64] > 0.5f);
      const bool ym = (y > 0)  && (sm[SLAB + k - 64] > 0.5f);
      const bool xm = (x > 0)  && (sm[SLAB + k - 1]  > 0.5f);
      const bool xp = (x < 63) && (sm[SLAB + k + 1]  > 0.5f);
      // c order matches _SHIFTS: (-z,+z,+y,-y,-x,+x)
      m = (u32)(!zm) | ((u32)(!zp) << 1) | ((u32)(!yp) << 2) |
          ((u32)(!ym) << 3) | ((u32)(!xm) << 4) | ((u32)(!xp) << 5);
    }
    mask[mb + k] = (u8)m;
    lc0 += m & 1u;        lc1 += (m >> 1) & 1u; lc2 += (m >> 2) & 1u;
    lc3 += (m >> 3) & 1u; lc4 += (m >> 4) & 1u; lc5 += (m >> 5) & 1u;
  }
  atomicAdd(&cnt[0], lc0); atomicAdd(&cnt[1], lc1); atomicAdd(&cnt[2], lc2);
  atomicAdd(&cnt[3], lc3); atomicAdd(&cnt[4], lc4); atomicAdd(&cnt[5], lc5);
  __syncthreads();
  if (tid < 6) ws[W_SLABCNT + (b * 6 + tid) * GD + z] = cnt[tid];
}

// -------------------------------- WMMA IU8 byte-sum: per-batch quad totals
// A = per-byte popcounts of face masks (16x64 u8), B = ones -> D rows = row sums.
__global__ void k_count_wmma(u32* __restrict__ ws) {
  const int wid  = (int)(blockIdx.x * blockDim.x + threadIdx.x) >> 5;  // 512 waves
  const int lane = threadIdx.x & 31;                                   // wave32
  const u8* mask = (const u8*)ws;
  const size_t base = (size_t)wid * 1024;
  v8i a = *(const v8i*)(mask + base + (size_t)lane * 32);
  #pragma unroll
  for (int i = 0; i < 8; ++i) {        // SWAR per-byte popcount (values 0..6)
    u32 m = (u32)a[i];
    m = m - ((m >> 1) & 0x55555555u);
    m = (m & 0x33333333u) + ((m >> 2) & 0x33333333u);
    m = (m + (m >> 4)) & 0x0F0F0F0Fu;
    a[i] = (int)m;
  }
  v8i ones;
  #pragma unroll
  for (int i = 0; i < 8; ++i) ones[i] = 0x01010101;
  v8i cz = {0, 0, 0, 0, 0, 0, 0, 0};
  v8i d = __builtin_amdgcn_wmma_i32_16x16x64_iu8(false, a, false, ones, cz,
                                                 false, false);
  int s = 0;
  #pragma unroll
  for (int i = 0; i < 8; ++i) s += d[i];
  const int r0  = __shfl(s, 0, 32);    // column 0: rows 0..7
  const int r16 = __shfl(s, 16, 32);   // column 0: rows 8..15
  if (lane == 0) {
    const int b = (base < (size_t)(GD * SLAB)) ? 0 : 1;
    atomicAdd(&ws[W_QPERB + b], (u32)(r0 + r16));
  }
}

// ----------------------------------------- exclusive scan of 768 slab counters
__global__ void k_scan_slabs(u32* __restrict__ ws) {
  if (threadIdx.x == 0) {
    u32 run = 0;
    for (int i = 0; i < 768; ++i) { ws[W_SLABBASE + i] = run; run += ws[W_SLABCNT + i]; }
    ws[W_SLABBASE + 768] = run;      // total quads
  }
}

// ------------------- quad emission + vertex bitmap + directed edge masks
__global__ void k_emit_quads(u32* __restrict__ ws) {
  __shared__ u32 tot[256];
  const int sb = blockIdx.x;                       // (b,c,z) in reference order
  const int b = sb / 384, c = (sb / 64) % 6, z = sb % 64;
  const int tid = threadIdx.x;
  const u8* mask = (const u8*)ws;
  const size_t cbase = (size_t)(b * GD + z) * SLAB;

  u8 loc[16]; u32 cnt = 0;
  for (int j = 0; j < 16; ++j) {                   // cells in (y,x) order
    const int k = tid * 16 + j;
    loc[j] = (mask[cbase + k] >> c) & 1u;
    cnt += loc[j];
  }
  tot[tid] = cnt;
  __syncthreads();
  u32 myb = ws[W_SLABBASE + sb];
  for (int i = 0; i < tid; ++i) myb += tot[i];

  const int ea[5]  = {0, 1, 0, 2, 0};              // tri edges (0,1)(1,2)(0,2)(2,3)(0,3)
  const int eb2[5] = {1, 2, 2, 3, 3};
  u32 run = 0;
  for (int j = 0; j < 16; ++j) {
    if (!loc[j]) continue;
    const int k = tid * 16 + j;
    const int y = k >> 6, x = k & 63;
    const u32 q = myb + (run++);
    ws[W_QUADS + q] = (u32)x | ((u32)y << 6) | ((u32)z << 12) |
                      ((u32)c << 18) | ((u32)b << 21);
    int pc[4]; u8 cb[4];
    #pragma unroll
    for (int kk = 0; kk < 4; ++kk) {
      const u8 ct = cornerTab[c][kk];
      cb[kk] = ct;
      const int lz = z + ((ct >> 2) & 1), ly = y + ((ct >> 1) & 1), lx = x + (ct & 1);
      const int p = (lz * L + ly) * L + lx;
      pc[kk] = p;
      atomicOr(&ws[W_VBITMAP + b * WPB + (p >> 5)], 1u << (p & 31));
    }
    #pragma unroll
    for (int e = 0; e < 5; ++e) {
      const int a2 = ea[e], b2 = eb2[e];
      const int ddz = ((cb[b2] >> 2) & 1) - ((cb[a2] >> 2) & 1);
      const int ddy = ((cb[b2] >> 1) & 1) - ((cb[a2] >> 1) & 1);
      const int ddx = (cb[b2] & 1) - (cb[a2] & 1);
      const int idxf = (ddz + 1) * 9 + (ddy + 1) * 3 + (ddx + 1);
      const int idxr = (1 - ddz) * 9 + (1 - ddy) * 3 + (1 - ddx);
      atomicOr(&ws[W_EDGEMASK + b * LSIZE + pc[a2]], 1u << idxf);
      atomicOr(&ws[W_EDGEMASK + b * LSIZE + pc[b2]], 1u << idxr);
    }
  }
}

// --------------- rank prefixes over vertex bitmap + edge-degree weights
__global__ void k_scan_verts(u32* __restrict__ ws) {
  __shared__ u32 part[256];
  const int t = threadIdx.x;
  const int per = (TW + 255) / 256;    // 68 words/thread, in order
  // pass 1: vertex ranks (lattice order == np.unique sorted order)
  u32 s = 0;
  for (int i = 0; i < per; ++i) {
    const int w = t * per + i;
    if (w < TW) s += __popc(ws[W_VBITMAP + w]);
  }
  part[t] = s; __syncthreads();
  u32 run = 0;
  for (int i = 0; i < t; ++i) run += part[i];
  for (int i = 0; i < per; ++i) {
    const int w = t * per + i;
    if (w < TW) { ws[W_WORDPREF + w] = run; run += __popc(ws[W_VBITMAP + w]); }
  }
  if (t == 255) ws[W_WORDPREF + TW] = run;
  __syncthreads();
  if (t == 0) {
    ws[W_VPERB]     = ws[W_WORDPREF + WPB];
    ws[W_VPERB + 1] = ws[W_WORDPREF + TW] - ws[W_WORDPREF + WPB];
    ws[W_VPERB + 2] = ws[W_WORDPREF + TW];
  }
  __syncthreads();
  // pass 2: edge ranks (weight = popcount of neighbor mask per set vertex)
  u32 s2 = 0;
  for (int i = 0; i < per; ++i) {
    const int w = t * per + i;
    if (w < TW) {
      u32 bits = ws[W_VBITMAP + w];
      const int b = (w >= WPB), wofs = (w - b * WPB) * 32;
      while (bits) {
        const int j = __ffs(bits) - 1; bits &= bits - 1;
        s2 += __popc(ws[W_EDGEMASK + b * LSIZE + wofs + j]);
      }
    }
  }
  part[t] = s2; __syncthreads();
  u32 run2 = 0;
  for (int i = 0; i < t; ++i) run2 += part[i];
  for (int i = 0; i < per; ++i) {
    const int w = t * per + i;
    if (w < TW) {
      ws[W_EWORDPREF + w] = run2;
      u32 bits = ws[W_VBITMAP + w];
      const int b = (w >= WPB), wofs = (w - b * WPB) * 32;
      while (bits) {
        const int j = __ffs(bits) - 1; bits &= bits - 1;
        run2 += __popc(ws[W_EDGEMASK + b * LSIZE + wofs + j]);
      }
    }
  }
  if (t == 255) ws[W_EWORDPREF + TW] = run2;
}

__device__ __forceinline__ u32 vrank(const u32* ws, int b, int p) {
  const int w = b * WPB + (p >> 5);
  const u32 below = ws[W_VBITMAP + w] & ((1u << (p & 31)) - 1u);
  return ws[W_WORDPREF + w] + __popc(below);
}

// ------------------------------------------------------- vs (+ v_index) output
__global__ void k_emit_vertices(const u32* __restrict__ ws, u32* __restrict__ out) {
  const int w = blockIdx.x * 256 + threadIdx.x;
  if (w == 0) {                       // v_index: two int64 after Nv*3 floats
    const u32 Nv = ws[W_VPERB + 2];
    out[3 * (size_t)Nv]     = ws[W_VPERB];     out[3 * (size_t)Nv + 1] = 0;
    out[3 * (size_t)Nv + 2] = ws[W_VPERB + 1]; out[3 * (size_t)Nv + 3] = 0;
  }
  if (w >= TW) return;
  u32 bits = ws[W_VBITMAP + w];
  u32 id = ws[W_WORDPREF + w];
  const int b = (w >= WPB), wofs = (w - b * WPB) * 32;
  float* fo = (float*)out;
  while (bits) {
    const int j = __ffs(bits) - 1; bits &= bits - 1;
    const int p = wofs + j;
    const int lz = p / LL, r = p % LL, ly = r / L, lx = r % L;
    fo[(size_t)id * 3 + 0] = (float)lz - 0.5f;
    fo[(size_t)id * 3 + 1] = (float)ly - 0.5f;
    fo[(size_t)id * 3 + 2] = (float)lx - 0.5f;
    ++id;
  }
}

// ------------------------------------------------- faces_local (+ f_index)
__global__ void k_emit_faces(const u32* __restrict__ ws, u32* __restrict__ out) {
  const u32 total = ws[W_SLABBASE + 768];
  const u32 Nv = ws[W_VPERB + 2];
  const size_t FB = 3 * (size_t)Nv + 4;
  const u32 q = blockIdx.x * 256 + threadIdx.x;
  if (q == 0) {                        // f_index = 2 * quads per batch (WMMA totals)
    const size_t fo = FB + (size_t)total * 12;
    out[fo]     = 2u * ws[W_QPERB];     out[fo + 1] = 0;
    out[fo + 2] = 2u * ws[W_QPERB + 1]; out[fo + 3] = 0;
  }
  if (q >= total) return;
  const u32 pk = ws[W_QUADS + q];
  const int x = pk & 63, y = (pk >> 6) & 63, z = (pk >> 12) & 63;
  const int c = (pk >> 18) & 7, b = (pk >> 21) & 1;
  const u32 voff = b ? ws[W_VPERB] : 0u;
  u32 ids[4];
  #pragma unroll
  for (int kk = 0; kk < 4; ++kk) {
    const u8 ct = cornerTab[c][kk];
    const int p = ((z + ((ct >> 2) & 1)) * L + (y + ((ct >> 1) & 1))) * L + (x + (ct & 1));
    ids[kk] = vrank(ws, b, p) - voff;
  }
  const size_t o = FB + (size_t)q * 12;      // 2 triangles x 3 int64 ids
  out[o + 0]  = ids[0]; out[o + 1]  = 0;
  out[o + 2]  = ids[1]; out[o + 3]  = 0;
  out[o + 4]  = ids[2]; out[o + 5]  = 0;
  out[o + 6]  = ids[0]; out[o + 7]  = 0;
  out[o + 8]  = ids[2]; out[o + 9]  = 0;
  out[o + 10] = ids[3]; out[o + 11] = 0;
}

// ----------------------------------------------------------- adjacency [2,Ne]
__global__ void k_emit_adj(const u32* __restrict__ ws, u32* __restrict__ out) {
  const int w = blockIdx.x * 256 + threadIdx.x;
  if (w >= TW) return;
  const u32 total = ws[W_SLABBASE + 768];
  const u32 Nv = ws[W_VPERB + 2];
  const u32 Ne = ws[W_EWORDPREF + TW];
  const size_t AB = 3 * (size_t)Nv + 4 + (size_t)total * 12 + 4;
  u32 bits = ws[W_VBITMAP + w];
  u32 uid = ws[W_WORDPREF + w];
  u32 eb = ws[W_EWORDPREF + w];
  const int b = (w >= WPB), wofs = (w - b * WPB) * 32;
  while (bits) {
    const int j = __ffs(bits) - 1; bits &= bits - 1;
    const int p = wofs + j;
    const u32 em = ws[W_EDGEMASK + b * LSIZE + p];
    for (int idx = 0; idx < 27; ++idx) {       // ascending idx == ascending neighbor id
      if ((em >> idx) & 1u) {
        const int p2 = p + (idx / 9 - 1) * LL + ((idx / 3) % 3 - 1) * L + (idx % 3 - 1);
        const u32 nid = vrank(ws, b, p2);
        out[AB + 2 * (size_t)eb]        = uid; out[AB + 2 * (size_t)eb + 1] = 0;
        out[AB + 2 * ((size_t)Ne + eb)] = nid; out[AB + 2 * ((size_t)Ne + eb) + 1] = 0;
        ++eb;
      }
    }
    ++uid;
  }
}

extern "C" void kernel_launch(void* const* d_in, const int* in_sizes, int n_in,
                              void* d_out, int out_size, void* d_ws, size_t ws_size,
                              hipStream_t stream) {
  (void)in_sizes; (void)n_in; (void)out_size; (void)ws_size;
  const float* t = (const float*)d_in[0];
  u32* ws = (u32*)d_ws;
  u32* out = (u32*)d_out;

  k_zero         <<<(ZTOT + 255) / 256, 256, 0, stream>>>(ws);
  k_stencil      <<<NB * GD,            256, 0, stream>>>(t, ws);
  k_count_wmma   <<<64,                 256, 0, stream>>>(ws);   // 512 wave32 waves
  k_scan_slabs   <<<1,                  256, 0, stream>>>(ws);
  k_emit_quads   <<<NB * 6 * GD,        256, 0, stream>>>(ws);
  k_scan_verts   <<<1,                  256, 0, stream>>>(ws);
  k_emit_vertices<<<(TW + 255) / 256,   256, 0, stream>>>(ws, out);
  k_emit_faces   <<<12288,              256, 0, stream>>>(ws, out);  // 6*N/256 worst case
  k_emit_adj     <<<(TW + 255) / 256,   256, 0, stream>>>(ws, out);
}
